// Evoformer_81655918231695
// MI455X (gfx1250) — matrix operand order repaired and checked by reading, not compiled, not run on hardware
//
#include <hip/hip_runtime.h>
#include <hip/hip_bf16.h>

// ---------------------------------------------------------------------------
// Evoformer block for MI455X (gfx1250, wave32). All matmuls go through
// v_wmma_f32_16x16x32_f16 (f16 inputs, f32 accumulation). Attention and
// outer-product-mean are fused so the large score / outer tensors never hit
// memory; K tiles are staged with CDNA5 async global->LDS copies.
// ---------------------------------------------------------------------------

typedef _Float16 v16h __attribute__((ext_vector_type(16)));
typedef float    v8f  __attribute__((ext_vector_type(8)));

static __device__ inline v8f wmma16(const v16h& a, const v16h& b, const v8f& c) {
  return __builtin_amdgcn_wmma_f32_16x16x32_f16(false, a, false, b, (short)0, c, false, false);
}

// ------------------------------ LayerNorm ----------------------------------
__global__ void ln_kernel(const float* __restrict__ x, const float* __restrict__ g,
                          const float* __restrict__ b, float* __restrict__ y,
                          int rows, int C) {
  int wave = threadIdx.x >> 5, lane = threadIdx.x & 31;
  int row = blockIdx.x * (blockDim.x >> 5) + wave;
  if (row >= rows) return;
  const float* xr = x + (long)row * C;
  float s = 0.f, s2 = 0.f;
  for (int c = lane; c < C; c += 32) { float t = xr[c]; s += t; s2 += t * t; }
  for (int m = 16; m; m >>= 1) { s += __shfl_xor(s, m, 32); s2 += __shfl_xor(s2, m, 32); }
  float mu = s / C;
  float var = s2 / C - mu * mu;
  float inv = rsqrtf(var + 1e-5f);
  float* yr = y + (long)row * C;
  for (int c = lane; c < C; c += 32) yr[c] = (xr[c] - mu) * inv * g[c] + b[c];
}

// ------------------------------ Generic GEMM -------------------------------
// C[b,m,n] = epi(A op B + bias) (+ residual). A staged [64][32] f16 in LDS,
// B staged transposed [n][k] so WMMA B fragments read 16 contiguous halfs.
template <int EPI>  // 0=none 1=relu 2=sigmoid
__global__ __launch_bounds__(128)
void gemm_kernel(const float* __restrict__ A, int lda, long sAb, int tA,
                 const float* __restrict__ B, int ldb, long sBb, int tB,
                 const float* __restrict__ bias,
                 const float* __restrict__ Rr, float* __restrict__ Cp,
                 long sCb, long scm, int scn, int M, int N, int K) {
  __shared__ _Float16 As[64 * 36];
  __shared__ _Float16 Bs[64 * 36];
  int batch = blockIdx.z;
  A += (long)batch * sAb;
  B += (long)batch * sBb;
  long cbase = (long)batch * sCb;
  int m0 = blockIdx.y * 64, n0 = blockIdx.x * 64;
  int tid = threadIdx.x, wave = tid >> 5, lane = tid & 31;
  int wm = wave >> 1, wn = wave & 1;
  int half = lane >> 4, l15 = lane & 15;

  v8f vzero = {};
  v8f acc[2][2];
  acc[0][0] = vzero; acc[0][1] = vzero; acc[1][0] = vzero; acc[1][1] = vzero;

  for (int k0 = 0; k0 < K; k0 += 32) {
    for (int t = tid; t < 64 * 32; t += 128) {
      int mm = t >> 5, kk = t & 31;
      int gm = m0 + mm, gk = k0 + kk;
      float va = 0.f;
      if (gm < M) va = A[tA ? ((long)gk * lda + gm) : ((long)gm * lda + gk)];
      As[mm * 36 + kk] = (_Float16)va;
      int gn = n0 + mm;
      float vb = 0.f;
      if (gn < N) vb = B[tB ? ((long)gn * ldb + gk) : ((long)gk * ldb + gn)];
      Bs[mm * 36 + kk] = (_Float16)vb;
    }
    __syncthreads();
    v16h af[2], bf[2];
#pragma unroll
    for (int f = 0; f < 2; f++) {
      const _Float16* pa = &As[(wm * 32 + f * 16 + l15) * 36 + half * 16];
#pragma unroll
      for (int e = 0; e < 16; e++) af[f][e] = pa[e];
      const _Float16* pb = &Bs[(wn * 32 + f * 16 + l15) * 36 + half * 16];
#pragma unroll
      for (int e = 0; e < 16; e++) bf[f][e] = pb[e];
    }
#pragma unroll
    for (int i = 0; i < 2; i++)
#pragma unroll
      for (int j = 0; j < 2; j++) acc[i][j] = wmma16(af[i], bf[j], acc[i][j]);
    __syncthreads();
  }

#pragma unroll
  for (int i = 0; i < 2; i++)
#pragma unroll
    for (int j = 0; j < 2; j++) {
      int ng = n0 + wn * 32 + j * 16 + l15;
      if (ng >= N) continue;
      float bv = bias ? bias[ng] : 0.f;
      long addr[8];
      float rv[8];
#pragma unroll
      for (int r = 0; r < 8; r++) {
        int mg = m0 + wm * 32 + i * 16 + r + half * 8;
        addr[r] = (mg < M) ? (cbase + (long)mg * scm + (long)ng * scn) : -1;
        rv[r] = 0.f;
      }
      if (Rr) {  // batched residual loads: issue as one clause, then use
#pragma unroll
        for (int r = 0; r < 8; r++)
          if (addr[r] >= 0) rv[r] = Rr[addr[r]];
      }
#pragma unroll
      for (int r = 0; r < 8; r++) {
        if (addr[r] < 0) continue;
        float v = acc[i][j][r] + bv;
        if (EPI == 1) v = fmaxf(v, 0.f);
        else if (EPI == 2) v = 1.f / (1.f + __expf(-v));
        Cp[addr[r]] = v + rv[r];
      }
    }
}

// ------------------------- Fused multi-head attention ----------------------
// q/k/v/o element (bh_hi,row,bh_lo,d) at  bh_hi*S_HI + row*Rst + bh_lo*32 + d.
// bias (optional): addr = qrow*bias_si + kcol*bias_sj + (bh%H).
// Block: 128 threads (4 waves, 16 queries each). K tile async-staged raw f32
// into LDS (CDNA5 global_load_async_to_lds_b128, ASYNCcnt), converted once to
// f16; scores in per-wave LDS (f16); softmax via wave shfl; P@V via WMMA.
template <int HAS_BIAS>
__global__ __launch_bounds__(128)
void attn_kernel(const float* __restrict__ q, const float* __restrict__ k,
                 const float* __restrict__ v, float* __restrict__ o,
                 const float* __restrict__ bias, int bias_si, int bias_sj,
                 int L, long S_HI, long Rst, int H, float scale) {
  __shared__ _Float16 kt[256 * 36];        // K tile  [L][32] f16
  __shared__ _Float16 sc[4 * 16 * 264];    // per-wave scores/probs [16][L]
  int tid = threadIdx.x, wave = tid >> 5, lane = tid & 31;
  int half = lane >> 4, l15 = lane & 15;
  int bh = blockIdx.y;
  int blo = bh % H;
  long base = (long)(bh / H) * S_HI + (long)blo * 32;

  // --- stage raw f32 K tile with async global->LDS (16B per lane op) ---
  {
    float* raw = reinterpret_cast<float*>(sc);  // reuse score region (32KB max)
    unsigned lds0 = (unsigned)(uintptr_t)raw;   // flat->LDS byte offset (low 32b)
    const char* kb = reinterpret_cast<const char*>(k + base);
    long rowbytes = Rst * 4;
    int nch = L * 8;  // 16B chunks (32 floats = 8 chunks per row)
    for (int cidx = tid; cidx < nch; cidx += 128) {
      int r = cidx >> 3, off = (cidx & 7) * 16;
      unsigned dst = lds0 + (unsigned)cidx * 16u;
      const void* src = kb + (long)r * rowbytes + off;
      asm volatile("global_load_async_to_lds_b128 %0, %1, off"
                   :: "v"(dst), "v"(src) : "memory");
    }
    asm volatile("s_wait_asynccnt 0x0" ::: "memory");
    __syncthreads();
    for (int t = tid; t < L * 32; t += 128) {
      int r = t >> 5, d = t & 31;
      kt[r * 36 + d] = (_Float16)raw[t];
    }
  }
  __syncthreads();

  int q0 = blockIdx.x * 64 + wave * 16;
  if (q0 < L) {
    v16h aq;
    {
      long qb = base + (long)(q0 + l15) * Rst;
#pragma unroll
      for (int e = 0; e < 16; e++) {
        int d = e + half * 16;
        aq[e] = (_Float16)(q[qb + d] * scale);
      }
    }
    _Float16* scw = &sc[wave * 16 * 264];
    for (int jt = 0; jt < (L >> 4); jt++) {
      v16h bk;
      const _Float16* pb = &kt[(jt * 16 + l15) * 36 + half * 16];
#pragma unroll
      for (int e = 0; e < 16; e++) bk[e] = pb[e];
      v8f s = {};
      s = wmma16(aq, bk, s);
      int col = jt * 16 + l15;
      float bvals[8];
      if (HAS_BIAS) {  // batched bias loads (one clause)
#pragma unroll
        for (int r = 0; r < 8; r++)
          bvals[r] = bias[(long)(q0 + r + half * 8) * bias_si + (long)col * bias_sj + blo];
      }
#pragma unroll
      for (int r = 0; r < 8; r++) {
        int m = r + half * 8;
        float val = s[r];
        if (HAS_BIAS) val += bvals[r];
        scw[m * 264 + col] = (_Float16)val;
      }
    }
    // row-wise softmax (per-wave data, no barrier needed)
    for (int m = 0; m < 16; m++) {
      _Float16* row = &scw[m * 264];
      float mx = -1e30f;
      for (int j = lane; j < L; j += 32) mx = fmaxf(mx, (float)row[j]);
      for (int msk = 16; msk; msk >>= 1) mx = fmaxf(mx, __shfl_xor(mx, msk, 32));
      float sum = 0.f;
      for (int j = lane; j < L; j += 32) {
        float e = __expf((float)row[j] - mx);
        row[j] = (_Float16)e;
        sum += e;
      }
      for (int msk = 16; msk; msk >>= 1) sum += __shfl_xor(sum, msk, 32);
      float inv = 1.f / sum;
      for (int j = lane; j < L; j += 32) row[j] = (_Float16)((float)row[j] * inv);
    }
    // o = P @ V
    v8f oa0 = {}, oa1 = {};
    for (int ks = 0; ks < (L >> 5); ks++) {
      v16h ap;
      const _Float16* pa = &scw[l15 * 264 + ks * 32 + half * 16];
#pragma unroll
      for (int e = 0; e < 16; e++) ap[e] = pa[e];
      int jrow = ks * 32 + half * 16;
#pragma unroll
      for (int fn = 0; fn < 2; fn++) {
        v16h bv;
        long vb = base + fn * 16 + l15;
#pragma unroll
        for (int e = 0; e < 16; e++) bv[e] = (_Float16)v[vb + (long)(jrow + e) * Rst];
        if (fn == 0) oa0 = wmma16(ap, bv, oa0);
        else         oa1 = wmma16(ap, bv, oa1);
      }
    }
#pragma unroll
    for (int r = 0; r < 8; r++) {
      int m = r + half * 8;
      long ob = base + (long)(q0 + m) * Rst + l15;
      o[ob] = oa0[r];
      o[ob + 16] = oa1[r];
    }
  }
}

// ------------------- Fused outer-product-mean + projection -----------------
// Block (jt, i): O[c,(j,d)] = (1/S) * sum_s a[s,i,c] b[s,j,d]  (32 x 512, f16
// in LDS via WMMA, K=S=64), then pair[i,j,:] += vec(O) @ wo + bo via WMMA.
__global__ __launch_bounds__(128)
void opm_kernel(const float* __restrict__ a, const float* __restrict__ b,
                const float* __restrict__ wo, const float* __restrict__ bo,
                float* __restrict__ pair, int Nn, float invS) {
  __shared__ _Float16 Aa[32 * 72];    // a_i^T [32 c][64 s]
  __shared__ _Float16 Ol[32 * 520];   // O [32 c][512 (j,d)]
  int i = blockIdx.y, jt = blockIdx.x;
  int tid = threadIdx.x, wave = tid >> 5, lane = tid & 31;
  int half = lane >> 4, l15 = lane & 15;

  for (int t = tid; t < 32 * 64; t += 128) {
    int c = t >> 6, s = t & 63;
    Aa[c * 72 + s] = (_Float16)(a[(long)s * 8192 + i * 32 + c] * invS);
  }
  __syncthreads();

  {  // step 1: each wave computes a 128-wide column chunk of O
    v8f vzero = {};
    v8f acc[2][8];
#pragma unroll
    for (int f = 0; f < 2; f++)
#pragma unroll
      for (int fn = 0; fn < 8; fn++) acc[f][fn] = vzero;
    for (int ks = 0; ks < 2; ks++) {
      v16h af[2];
#pragma unroll
      for (int f = 0; f < 2; f++) {
        const _Float16* pa = &Aa[(f * 16 + l15) * 72 + ks * 32 + half * 16];
#pragma unroll
        for (int e = 0; e < 16; e++) af[f][e] = pa[e];
      }
      int srow = ks * 32 + half * 16;
#pragma unroll
      for (int fn = 0; fn < 8; fn++) {
        int qc = wave * 128 + fn * 16 + l15;
        v16h bf;
#pragma unroll
        for (int e = 0; e < 16; e++)
          bf[e] = (_Float16)b[(long)(srow + e) * 8192 + jt * 512 + qc];
        acc[0][fn] = wmma16(af[0], bf, acc[0][fn]);
        acc[1][fn] = wmma16(af[1], bf, acc[1][fn]);
      }
    }
#pragma unroll
    for (int f = 0; f < 2; f++)
#pragma unroll
      for (int fn = 0; fn < 8; fn++)
#pragma unroll
        for (int r = 0; r < 8; r++) {
          int rc = f * 16 + r + half * 8;
          int col = wave * 128 + fn * 16 + l15;
          Ol[rc * 520 + col] = (_Float16)acc[f][fn][r];
        }
  }
  __syncthreads();

  {  // step 2: out[16 j][128 z]; wave owns 32 z-columns; K = 1024
    v8f acc0 = {}, acc1 = {};
    for (int ks = 0; ks < 32; ks++) {
      v16h ap;
      const _Float16* pa = &Ol[ks * 520 + l15 * 32 + half * 16];
#pragma unroll
      for (int e = 0; e < 16; e++) ap[e] = pa[e];
      int qrow = ks * 32 + half * 16;
#pragma unroll
      for (int fn = 0; fn < 2; fn++) {
        int z = wave * 32 + fn * 16 + l15;
        v16h bw;
#pragma unroll
        for (int e = 0; e < 16; e++) bw[e] = (_Float16)wo[(long)(qrow + e) * 128 + z];
        if (fn == 0) acc0 = wmma16(ap, bw, acc0);
        else         acc1 = wmma16(ap, bw, acc1);
      }
    }
#pragma unroll
    for (int r = 0; r < 8; r++) {
      int j = jt * 16 + r + half * 8;
#pragma unroll
      for (int fn = 0; fn < 2; fn++) {
        int z = wave * 32 + fn * 16 + l15;
        long addr = ((long)i * Nn + j) * 128 + z;
        pair[addr] = pair[addr] + (fn == 0 ? acc0[r] : acc1[r]) + bo[z];
      }
    }
  }
}

// ------------------------------ Small kernels ------------------------------
__global__ void bias_proj_kernel(const float* __restrict__ z, const float* __restrict__ w,
                                 float* __restrict__ out, long rows, int Cin, int H) {
  long idx = (long)blockIdx.x * 256 + threadIdx.x;
  if (idx >= rows * H) return;
  int h = (int)(idx % H);
  long r = idx / H;
  const float* zr = z + r * Cin;
  float s = 0.f;
  for (int c = 0; c < Cin; c++) s += zr[c] * w[c * H + h];
  out[idx] = s;
}

__global__ void mul_kernel(float* o, const float* a, const float* b, long n) {
  long i = (long)blockIdx.x * 256 + threadIdx.x;
  if (i < n) o[i] = a[i] * b[i];
}

__global__ void fma_kernel(float* o, const float* res, const float* g, const float* t, long n) {
  long i = (long)blockIdx.x * 256 + threadIdx.x;
  if (i < n) o[i] = res[i] + g[i] * t[i];
}

// [N,N,C] -> [C,N,N]
__global__ void perm_cnn_kernel(float* __restrict__ out, const float* __restrict__ in,
                                int N, int C) {
  long t = (long)blockIdx.x * 256 + threadIdx.x;
  long total = (long)N * N * C;
  if (t >= total) return;
  int c = (int)(t % C);
  long ij = t / C;
  int j = (int)(ij % N), i = (int)(ij / N);
  out[(long)c * N * N + (long)i * N + j] = in[t];
}

// out[j,i,c] = in[i,j,c]
__global__ void transpose_nnc_kernel(float* __restrict__ out, const float* __restrict__ in,
                                     int N, int C) {
  long t = (long)blockIdx.x * 256 + threadIdx.x;
  long total = (long)N * N * C;
  if (t >= total) return;
  int c = (int)(t % C);
  long ij = t / C;
  int j = (int)(ij % N), i = (int)(ij / N);
  out[((long)j * N + i) * C + c] = in[t];
}

// pair[i,j,c] += src[j,i,c]
__global__ void transpose_add_kernel(float* __restrict__ pair, const float* __restrict__ src,
                                     int N, int C) {
  long t = (long)blockIdx.x * 256 + threadIdx.x;
  long total = (long)N * N * C;
  if (t >= total) return;
  int c = (int)(t % C);
  long ij = t / C;
  int j = (int)(ij % N), i = (int)(ij / N);
  pair[t] += src[((long)j * N + i) * C + c];
}

// ------------------------------- Host driver -------------------------------
extern "C" void kernel_launch(void* const* d_in, const int* in_sizes, int n_in,
                              void* d_out, int out_size, void* d_ws, size_t ws_size,
                              hipStream_t stream) {
  (void)in_sizes; (void)n_in; (void)out_size; (void)ws_size;
  const float SCALE = 0.17677669529663687f;  // 1/sqrt(32)
  const int N = 256;
  const long MSA_E = 64L * 256 * 256;    // 4,194,304
  const long PAIR_E = 256L * 256 * 128;  // 8,388,608

#define IN(i) ((const float*)d_in[i])
  const float* msa_in = IN(0);
  const float* pair_in = IN(1);
  float* succ = (float*)d_out;        // msa accumulator -> first output
  float* pairA = succ + MSA_E;        // pair accumulator -> second output
  float* W = (float*)d_ws;
  const long SLOT = 8388608;          // 8M floats per slot; needs ~170MB ws
  float *s0 = W, *s1 = W + SLOT, *s2 = W + 2 * SLOT, *s3 = W + 3 * SLOT, *s4 = W + 4 * SLOT;
  float* smallb = W + 5 * SLOT;       // up to 0.5M floats

  hipMemcpyAsync(succ, msa_in, MSA_E * sizeof(float), hipMemcpyDeviceToDevice, stream);
  hipMemcpyAsync(pairA, pair_in, PAIR_E * sizeof(float), hipMemcpyDeviceToDevice, stream);

  auto LN = [&](const float* x, const float* g, const float* b, float* y, long rows, int C) {
    ln_kernel<<<dim3((unsigned)(rows / 8)), 256, 0, stream>>>(x, g, b, y, (int)rows, C);
  };
  auto GEMMB = [&](int epi, const float* A, int lda, long sAb, int tA,
                   const float* B, int ldb, long sBb, int tB,
                   const float* bias, const float* Rr, float* Cp,
                   long sCb, long scm, int scn, int M, int Nn, int K, int batch) {
    dim3 g((Nn + 63) / 64, (M + 63) / 64, batch), blk(128);
    if (epi == 0)
      gemm_kernel<0><<<g, blk, 0, stream>>>(A, lda, sAb, tA, B, ldb, sBb, tB, bias, Rr, Cp, sCb, scm, scn, M, Nn, K);
    else if (epi == 1)
      gemm_kernel<1><<<g, blk, 0, stream>>>(A, lda, sAb, tA, B, ldb, sBb, tB, bias, Rr, Cp, sCb, scm, scn, M, Nn, K);
    else
      gemm_kernel<2><<<g, blk, 0, stream>>>(A, lda, sAb, tA, B, ldb, sBb, tB, bias, Rr, Cp, sCb, scm, scn, M, Nn, K);
  };
  auto GEMS = [&](int epi, const float* A, const float* B, const float* bias,
                  const float* Rr, float* Cp, int M, int Nn, int K) {
    GEMMB(epi, A, K, 0, 0, B, Nn, 0, 0, bias, Rr, Cp, 0, Nn, 1, M, Nn, K, 1);
  };
  auto EW = [&](long n) { return dim3((unsigned)((n + 255) / 256)); };

  // ---------------- MSA row attention (with pair bias) ----------------
  float* mn = s0;  float* qb = s0 + MSA_E;
  float* kb = s1;  float* vb = s1 + MSA_E;
  float* gb = s2;  float* ob = s2 + MSA_E;
  float* zn = s3;
  LN(succ, IN(2), IN(3), mn, 16384, 256);
  LN(pairA, IN(4), IN(5), zn, 65536, 128);
  bias_proj_kernel<<<EW(65536L * 8), 256, 0, stream>>>(zn, IN(13), smallb, 65536, 128, 8);
  GEMS(0, mn, IN(6), nullptr, nullptr, qb, 16384, 256, 256);
  GEMS(0, mn, IN(7), nullptr, nullptr, kb, 16384, 256, 256);
  GEMS(0, mn, IN(8), nullptr, nullptr, vb, 16384, 256, 256);
  GEMS(2, mn, IN(9), IN(10), nullptr, gb, 16384, 256, 256);
  attn_kernel<1><<<dim3(4, 512), 128, 0, stream>>>(qb, kb, vb, ob, smallb, 2048, 8,
                                                   256, 65536L, 256L, 8, SCALE);
  mul_kernel<<<EW(MSA_E), 256, 0, stream>>>(ob, ob, gb, MSA_E);
  GEMS(0, ob, IN(11), IN(12), succ, succ, 16384, 256, 256);

  // ---------------- MSA column attention ----------------
  LN(succ, IN(14), IN(15), mn, 16384, 256);
  GEMS(0, mn, IN(16), nullptr, nullptr, qb, 16384, 256, 256);
  GEMS(0, mn, IN(17), nullptr, nullptr, kb, 16384, 256, 256);
  GEMS(0, mn, IN(18), nullptr, nullptr, vb, 16384, 256, 256);
  GEMS(2, mn, IN(19), IN(20), nullptr, gb, 16384, 256, 256);
  attn_kernel<0><<<dim3(1, 2048), 128, 0, stream>>>(qb, kb, vb, ob, nullptr, 0, 0,
                                                    64, 256L, 65536L, 8, SCALE);
  mul_kernel<<<EW(MSA_E), 256, 0, stream>>>(ob, ob, gb, MSA_E);
  GEMS(0, ob, IN(21), IN(22), succ, succ, 16384, 256, 256);

  // ---------------- MSA transition ----------------
  LN(succ, IN(23), IN(24), mn, 16384, 256);
  float* hid = s1;  // 16M floats span s1..s2
  GEMS(1, mn, IN(25), IN(26), nullptr, hid, 16384, 1024, 256);
  GEMS(0, hid, IN(27), IN(28), succ, succ, 16384, 256, 1024);
  // succ now final (first output)

  // ---------------- Outer product mean ----------------
  float* mo = s0;
  LN(succ, IN(29), IN(30), mo, 16384, 256);
  float* opa = s1;
  float* opb = s1 + 524288;
  GEMS(0, mo, IN(31), IN(32), nullptr, opa, 16384, 32, 256);
  GEMS(0, mo, IN(33), IN(34), nullptr, opb, 16384, 32, 256);
  opm_kernel<<<dim3(16, 256), 128, 0, stream>>>(opa, opb, IN(35), IN(36), pairA, N, 1.f / 64.f);

  // ---------------- Triangle multiplications ----------------
  auto TRIMULT = [&](int p, bool outgoing) {
    LN(pairA, IN(p + 0), IN(p + 1), s0, 65536, 128);                  // zn
    GEMS(2, s0, IN(p + 4), IN(p + 5), nullptr, s1, 65536, 128, 128);  // sig gate a
    GEMS(0, s0, IN(p + 2), IN(p + 3), nullptr, s2, 65536, 128, 128);  // lin a
    mul_kernel<<<EW(PAIR_E), 256, 0, stream>>>(s1, s1, s2, PAIR_E);   // a -> s1
    GEMS(2, s0, IN(p + 8), IN(p + 9), nullptr, s2, 65536, 128, 128);  // sig gate b
    GEMS(0, s0, IN(p + 6), IN(p + 7), nullptr, s3, 65536, 128, 128);  // lin b
    mul_kernel<<<EW(PAIR_E), 256, 0, stream>>>(s2, s2, s3, PAIR_E);   // b -> s2
    GEMS(2, s0, IN(p + 12), IN(p + 13), nullptr, s3, 65536, 128, 128);// g -> s3
    perm_cnn_kernel<<<EW(PAIR_E), 256, 0, stream>>>(s0, s1, N, 128);  // ap -> s0
    perm_cnn_kernel<<<EW(PAIR_E), 256, 0, stream>>>(s4, s2, N, 128);  // bp -> s4
    // x[i,j,c] batched over c=128: outgoing A@B^T, incoming A^T@B
    GEMMB(0, s0, 256, 65536, outgoing ? 0 : 1,
          s4, 256, 65536, outgoing ? 1 : 0,
          nullptr, nullptr, s1, 1, (long)N * 128, 128, 256, 256, 256, 128);
    LN(s1, IN(p + 10), IN(p + 11), s2, 65536, 128);                   // xn -> s2
    GEMS(0, s2, IN(p + 14), IN(p + 15), nullptr, s0, 65536, 128, 128);// t -> s0
    fma_kernel<<<EW(PAIR_E), 256, 0, stream>>>(pairA, pairA, s3, s0, PAIR_E);
  };
  TRIMULT(37, true);   // outgoing
  TRIMULT(53, false);  // incoming

  // ---------------- Triangle attentions ----------------
  auto TRIATTN = [&](int p, bool endMode) {
    const float* zsrc = pairA;
    if (endMode) {
      transpose_nnc_kernel<<<EW(PAIR_E), 256, 0, stream>>>(s0, pairA, N, 128);
      zsrc = s0;
    }
    float* znl = endMode ? s1 : s0;
    LN(zsrc, IN(p + 8), IN(p + 9), znl, 65536, 128);
    float* qq = endMode ? s0 : s1;  // pt dead after LN in end mode
    GEMS(0, znl, IN(p + 0), nullptr, nullptr, qq, 65536, 128, 128);
    GEMS(0, znl, IN(p + 1), nullptr, nullptr, s2, 65536, 128, 128);
    GEMS(0, znl, IN(p + 2), nullptr, nullptr, s3, 65536, 128, 128);
    GEMS(2, znl, IN(p + 3), IN(p + 4), nullptr, s4, 65536, 128, 128);
    bias_proj_kernel<<<EW(65536L * 4), 256, 0, stream>>>(znl, IN(p + 7), smallb, 65536, 128, 4);
    float* oq = znl;  // zn dead
    attn_kernel<1><<<dim3(4, 1024), 128, 0, stream>>>(qq, s2, s3, oq, smallb, 1024, 4,
                                                      256, 32768L, 128L, 4, SCALE);
    mul_kernel<<<EW(PAIR_E), 256, 0, stream>>>(oq, oq, s4, PAIR_E);
    if (!endMode) {
      GEMS(0, oq, IN(p + 5), IN(p + 6), pairA, pairA, 65536, 128, 128);
    } else {
      GEMS(0, oq, IN(p + 5), IN(p + 6), nullptr, s2, 65536, 128, 128);
      transpose_add_kernel<<<EW(PAIR_E), 256, 0, stream>>>(pairA, s2, N, 128);
    }
  };
  TRIATTN(69, false);  // start
  TRIATTN(79, true);   // end

  // ---------------- Pair transition ----------------
  LN(pairA, IN(89), IN(90), s0, 65536, 128);
  float* hid2 = s1;  // 32M floats span s1..s4
  GEMS(1, s0, IN(91), IN(92), nullptr, hid2, 65536, 512, 128);
  GEMS(0, hid2, IN(93), IN(94), pairA, pairA, 65536, 128, 512);
#undef IN
}
